// LSTM_AlexGraves_29910152249712
// MI455X (gfx1250) — compile-verified, matrix operand order checked by reading, not run on hardware
//
#include <hip/hip_runtime.h>
#include <hip/hip_bf16.h>
#include <math.h>

// CDNA5 / gfx1250: wave32, WMMA 16x16x32 bf16 (f32 accum), LDS-staged B tiles.

typedef __attribute__((ext_vector_type(16))) __bf16 v16bf;
typedef __attribute__((ext_vector_type(8)))  __bf16 v8bf;
typedef __attribute__((ext_vector_type(8)))  float  v8f;
typedef __attribute__((ext_vector_type(4)))  int    v4i;

#define BATCH  8192
#define NSTEPS 16
#define IN_F   256
#define OUT_F  512
#define KTOT   768                   // 512 (H) + 256 (x)
#define LDS_STRIDE 392               // 384 + 8 pad -> 196 dwords (196%64==4): conflict-free

#if defined(__HIP_DEVICE_COMPILE__) && \
    __has_builtin(__builtin_amdgcn_global_load_async_to_lds_b128) && \
    __has_builtin(__builtin_amdgcn_s_wait_asynccnt)
#define USE_ASYNC 1
#else
#define USE_ASYNC 0
#endif

#if USE_ASYNC
// Builtin signature (from compiler diagnostic): param0 = v4i in addrspace(1)
// (printed as "int __vector(4) __device__ *"), param1 = LDS side, then
// imm offset and cpol.
typedef __attribute__((address_space(1))) v4i as1_v4i;
typedef __attribute__((address_space(3))) v4i as3_v4i;
#endif

__device__ __forceinline__ void copy16_g2lds(unsigned short* dst_lds,
                                             const unsigned short* src_g) {
#if USE_ASYNC
  __builtin_amdgcn_global_load_async_to_lds_b128(
      (as1_v4i*)src_g, (as3_v4i*)dst_lds, /*offset=*/0, /*cpol=*/0);
#else
  *(uint4*)dst_lds = *(const uint4*)src_g;
#endif
}

__device__ __forceinline__ v16bf combine8(v8bf lo, v8bf hi) {
  v16bf r;
#pragma unroll
  for (int i = 0; i < 8; ++i) { r[i] = lo[i]; r[i + 8] = hi[i]; }
  return r;
}

__device__ __forceinline__ v8f wmma_bf16(v16bf a, v16bf b, v8f c) {
  return __builtin_amdgcn_wmma_f32_16x16x32_bf16(
      false, a, false, b, (short)0, c, false, false);
}

__device__ __forceinline__ float sigmoidf_fast(float z) {
  return 1.0f / (1.0f + __expf(-z));
}

__device__ __forceinline__ unsigned int pk2(float a, float b) {
  union { __bf16 h[2]; unsigned int u; } un;
  un.h[0] = (__bf16)a; un.h[1] = (__bf16)b;
  return un.u;
}

// ---------------------------------------------------------------------------
// Kernel 0: pack f32 inputs into bf16 workspace.
//   Abf[b][0:512]=H[b], Abf[b][512:768]=x[b,step,:]           (8192 x 768)
//   Bbf[g*512+o][0:512]=W_h[g][o], [512:768]=W_x[g][o]        (2048 x 768)
// One 8-element (16B dst) chunk per thread; 768 % 8 == 0 and 512 % 8 == 0 so
// chunks never straddle the H/x boundary.
// ---------------------------------------------------------------------------
__global__ __launch_bounds__(256) void pack_bf16_kernel(
    const float* __restrict__ x, const float* __restrict__ H,
    const float* __restrict__ fg_w_h, const float* __restrict__ fg_w_x,
    const float* __restrict__ ig_w_h, const float* __restrict__ ig_w_x,
    const float* __restrict__ in_w_h, const float* __restrict__ in_w_x,
    const float* __restrict__ og_w_h, const float* __restrict__ og_w_x,
    const int* __restrict__ nstep,
    unsigned short* __restrict__ Abf, unsigned short* __restrict__ Bbf)
{
  const size_t c = (size_t)blockIdx.x * 256 + threadIdx.x;
  const size_t e = c * 8;
  const size_t A_elems = (size_t)BATCH * KTOT;
  const int step = *nstep;

  const float* src;
  unsigned short* dst;
  if (e < A_elems) {
    const int b = (int)(e / KTOT), j = (int)(e % KTOT);
    src = (j < OUT_F) ? (H + (size_t)b * OUT_F + j)
                      : (x + ((size_t)b * NSTEPS + step) * IN_F + (j - OUT_F));
    dst = Abf + e;
  } else {
    const size_t e2 = e - A_elems;
    const int row = (int)(e2 / KTOT), j = (int)(e2 % KTOT);
    const int g = row >> 9, o = row & (OUT_F - 1);
    const float* wh = g == 0 ? fg_w_h : g == 1 ? ig_w_h : g == 2 ? in_w_h : og_w_h;
    const float* wx = g == 0 ? fg_w_x : g == 1 ? ig_w_x : g == 2 ? in_w_x : og_w_x;
    src = (j < OUT_F) ? (wh + (size_t)o * OUT_F + j)
                      : (wx + (size_t)o * IN_F + (j - OUT_F));
    dst = Bbf + e2;
  }
  const float4* s4 = (const float4*)src;
  float4 f0 = s4[0], f1 = s4[1];
  uint4 p;
  p.x = pk2(f0.x, f0.y); p.y = pk2(f0.z, f0.w);
  p.z = pk2(f1.x, f1.y); p.w = pk2(f1.z, f1.w);
  *(uint4*)dst = p;
}

// ---------------------------------------------------------------------------
// Kernel 1: fused 4-gate GEMM + LSTM elementwise.
// Block = 256 threads = 8 waves. Block tile: 128 batch rows x 16 out features,
// all 4 gates. B tile (4x16xK) staged in LDS in two K-halves of 384.
// Wave w owns rows [mblk*128 + w*16, +16).
// ---------------------------------------------------------------------------
__global__ __launch_bounds__(256) void lstm_gemm_bf16(
    const unsigned short* __restrict__ Abf,
    const unsigned short* __restrict__ Bbf,
    const float* __restrict__ C,
    const float* __restrict__ fg_w_c, const float* __restrict__ fg_b,
    const float* __restrict__ ig_w_c, const float* __restrict__ ig_b,
    const float* __restrict__ in_b,
    const float* __restrict__ og_w_cn, const float* __restrict__ og_b,
    float* __restrict__ out)
{
  __shared__ unsigned short ldsB[64 * LDS_STRIDE];   // 4 gates x 16 rows, ~49KB

  const int lane = threadIdx.x & 31;
  const int wave = threadIdx.x >> 5;
  const int mblk = blockIdx.x >> 5;   // 0..63  (128-row batch block)
  const int nt   = blockIdx.x & 31;   // 0..31  (16-col output tile)
  const int half = lane >> 4;
  const int lm   = lane & 15;

  const int brow = mblk * 128 + wave * 16 + lm;  // this lane's A row
  const int o    = nt * 16 + lm;                 // output feature (B row)

  v8f acc_fg = {}, acc_ig = {}, acc_in = {}, acc_og = {};

  const __bf16* Ab = (const __bf16*)Abf;

#pragma unroll
  for (int kh = 0; kh < 2; ++kh) {
    const int kbase = kh * 384;
    if (kh) __syncthreads();   // all waves done reading previous half

    // Cooperative stage of B half: 64 rows x 384 bf16 = 3072 16B chunks.
#pragma unroll
    for (int t = 0; t < 12; ++t) {
      const int cchunk = threadIdx.x + t * 256;
      const int row = cchunk / 48;        // 0..63 : g*16 + orow
      const int cc  = cchunk % 48;        // 16B chunk within row
      const int g = row >> 4, orow = row & 15;
      const unsigned short* src =
          Bbf + (size_t)((g * OUT_F + nt * 16 + orow) * KTOT + kbase + cc * 8);
      unsigned short* dst = &ldsB[row * LDS_STRIDE + cc * 8];
      copy16_g2lds(dst, src);
    }
#if USE_ASYNC
    __builtin_amdgcn_s_wait_asynccnt(0);
#endif
    __syncthreads();

    // K loop over this half. Fragment layout: lane (m=L%16, half=L/16) holds
    // K=[k0+8*half,+8) in elems 0..7 and K=[k0+16+8*half,+8) in elems 8..15.
    const __bf16* arow = Ab + (size_t)brow * KTOT + kbase + half * 8;
    const __bf16* lb   = (const __bf16*)ldsB;
    const int lrow_off = lm * LDS_STRIDE + half * 8;

    for (int k0 = 0; k0 < 384; k0 += 32) {
      const v16bf a = combine8(*(const v8bf*)(arow + k0),
                               *(const v8bf*)(arow + k0 + 16));
      const __bf16* b0 = lb + (0 * 16 * LDS_STRIDE + lrow_off + k0);
      const __bf16* b1 = lb + (1 * 16 * LDS_STRIDE + lrow_off + k0);
      const __bf16* b2 = lb + (2 * 16 * LDS_STRIDE + lrow_off + k0);
      const __bf16* b3 = lb + (3 * 16 * LDS_STRIDE + lrow_off + k0);
      acc_fg = wmma_bf16(a, combine8(*(const v8bf*)b0, *(const v8bf*)(b0 + 16)), acc_fg);
      acc_ig = wmma_bf16(a, combine8(*(const v8bf*)b1, *(const v8bf*)(b1 + 16)), acc_ig);
      acc_in = wmma_bf16(a, combine8(*(const v8bf*)b2, *(const v8bf*)(b2 + 16)), acc_in);
      acc_og = wmma_bf16(a, combine8(*(const v8bf*)b3, *(const v8bf*)(b3 + 16)), acc_og);
    }
  }

  // Fused elementwise epilogue. D layout: n = lane%16, VGPR r -> m = r+8*half.
  const float wcf = fg_w_c[o], wci = ig_w_c[o], wco = og_w_cn[o];
  const float bbf = fg_b[o],  bbi = ig_b[o],  bbn = in_b[o], bbo = og_b[o];
  const int bb = mblk * 128 + wave * 16 + half * 8;

#pragma unroll
  for (int r = 0; r < 8; ++r) {
    const size_t idx = (size_t)(bb + r) * OUT_F + o;
    const float cv  = C[idx];
    const float fgv = sigmoidf_fast(acc_fg[r] + bbf + wcf * cv);
    const float igv = sigmoidf_fast(acc_ig[r] + bbi + wci * cv);
    const float inn = tanhf(acc_in[r] + bbn);
    const float cm  = fgv * cv + igv * inn;
    const float ogv = sigmoidf_fast(acc_og[r] + bbo + wco * cm);
    const float hv  = ogv * tanhf(cm);
    out[idx] = cm;
    out[(size_t)BATCH * OUT_F + idx] = hv;
  }
}

// ---------------------------------------------------------------------------
// Fallback (round-1 style): direct f32 loads + in-loop cvt; used only if the
// workspace is too small for the bf16 pack buffers.
// ---------------------------------------------------------------------------
__device__ __forceinline__ v16bf load_frag_f32(const float* __restrict__ p) {
  const float4* q0 = (const float4*)p;
  const float4* q1 = (const float4*)(p + 16);
  float4 f0 = q0[0], f1 = q0[1], f2 = q1[0], f3 = q1[1];
  v16bf r;
  r[0]  = (__bf16)f0.x; r[1]  = (__bf16)f0.y; r[2]  = (__bf16)f0.z; r[3]  = (__bf16)f0.w;
  r[4]  = (__bf16)f1.x; r[5]  = (__bf16)f1.y; r[6]  = (__bf16)f1.z; r[7]  = (__bf16)f1.w;
  r[8]  = (__bf16)f2.x; r[9]  = (__bf16)f2.y; r[10] = (__bf16)f2.z; r[11] = (__bf16)f2.w;
  r[12] = (__bf16)f3.x; r[13] = (__bf16)f3.y; r[14] = (__bf16)f3.z; r[15] = (__bf16)f3.w;
  return r;
}

__global__ __launch_bounds__(256) void lstm_wmma_f32_fallback(
    const float* __restrict__ x, const float* __restrict__ C, const float* __restrict__ H,
    const float* __restrict__ fg_w_c, const float* __restrict__ fg_w_h,
    const float* __restrict__ fg_w_x, const float* __restrict__ fg_b,
    const float* __restrict__ ig_w_c, const float* __restrict__ ig_w_h,
    const float* __restrict__ ig_w_x, const float* __restrict__ ig_b,
    const float* __restrict__ in_w_h, const float* __restrict__ in_w_x,
    const float* __restrict__ in_b,
    const float* __restrict__ og_w_cn, const float* __restrict__ og_w_h,
    const float* __restrict__ og_w_x,  const float* __restrict__ og_b,
    const int* __restrict__ nstep, float* __restrict__ out)
{
  const int lane = threadIdx.x & 31;
  const int wid  = (blockIdx.x << 3) + (threadIdx.x >> 5);
  const int mt = wid >> 5, nt = wid & 31;
  const int half = lane >> 4, lm = lane & 15;
  const int b0 = mt * 32 + lm, b1 = b0 + 16;
  const int o = nt * 16 + lm;
  const int step = *nstep;

  v8f c_fg0 = {}, c_fg1 = {}, c_ig0 = {}, c_ig1 = {};
  v8f c_in0 = {}, c_in1 = {}, c_og0 = {}, c_og1 = {};
  const int koff = half * 8;
  {
    const float* a0p = H + (size_t)b0 * OUT_F;
    const float* a1p = H + (size_t)b1 * OUT_F;
    const float* wf = fg_w_h + (size_t)o * OUT_F;
    const float* wi = ig_w_h + (size_t)o * OUT_F;
    const float* wn = in_w_h + (size_t)o * OUT_F;
    const float* wo = og_w_h + (size_t)o * OUT_F;
    for (int k0 = 0; k0 < OUT_F; k0 += 32) {
      const int k = k0 + koff;
      v16bf a0 = load_frag_f32(a0p + k), a1 = load_frag_f32(a1p + k), bm;
      bm = load_frag_f32(wf + k); c_fg0 = wmma_bf16(a0, bm, c_fg0); c_fg1 = wmma_bf16(a1, bm, c_fg1);
      bm = load_frag_f32(wi + k); c_ig0 = wmma_bf16(a0, bm, c_ig0); c_ig1 = wmma_bf16(a1, bm, c_ig1);
      bm = load_frag_f32(wn + k); c_in0 = wmma_bf16(a0, bm, c_in0); c_in1 = wmma_bf16(a1, bm, c_in1);
      bm = load_frag_f32(wo + k); c_og0 = wmma_bf16(a0, bm, c_og0); c_og1 = wmma_bf16(a1, bm, c_og1);
    }
  }
  {
    const float* a0p = x + ((size_t)b0 * NSTEPS + step) * IN_F;
    const float* a1p = x + ((size_t)b1 * NSTEPS + step) * IN_F;
    const float* wf = fg_w_x + (size_t)o * IN_F;
    const float* wi = ig_w_x + (size_t)o * IN_F;
    const float* wn = in_w_x + (size_t)o * IN_F;
    const float* wo = og_w_x + (size_t)o * IN_F;
    for (int k0 = 0; k0 < IN_F; k0 += 32) {
      const int k = k0 + koff;
      v16bf a0 = load_frag_f32(a0p + k), a1 = load_frag_f32(a1p + k), bm;
      bm = load_frag_f32(wf + k); c_fg0 = wmma_bf16(a0, bm, c_fg0); c_fg1 = wmma_bf16(a1, bm, c_fg1);
      bm = load_frag_f32(wi + k); c_ig0 = wmma_bf16(a0, bm, c_ig0); c_ig1 = wmma_bf16(a1, bm, c_ig1);
      bm = load_frag_f32(wn + k); c_in0 = wmma_bf16(a0, bm, c_in0); c_in1 = wmma_bf16(a1, bm, c_in1);
      bm = load_frag_f32(wo + k); c_og0 = wmma_bf16(a0, bm, c_og0); c_og1 = wmma_bf16(a1, bm, c_og1);
    }
  }
  const float wcf = fg_w_c[o], wci = ig_w_c[o], wco = og_w_cn[o];
  const float bbf = fg_b[o], bbi = ig_b[o], bbn = in_b[o], bbo = og_b[o];
#pragma unroll
  for (int t = 0; t < 2; ++t) {
    const v8f afg = t ? c_fg1 : c_fg0;
    const v8f aig = t ? c_ig1 : c_ig0;
    const v8f ain = t ? c_in1 : c_in0;
    const v8f aog = t ? c_og1 : c_og0;
    const int bb = mt * 32 + t * 16 + half * 8;
#pragma unroll
    for (int r = 0; r < 8; ++r) {
      const size_t idx = (size_t)(bb + r) * OUT_F + o;
      const float cv  = C[idx];
      const float fgv = sigmoidf_fast(afg[r] + bbf + wcf * cv);
      const float igv = sigmoidf_fast(aig[r] + bbi + wci * cv);
      const float inn = tanhf(ain[r] + bbn);
      const float cm  = fgv * cv + igv * inn;
      const float ogv = sigmoidf_fast(aog[r] + bbo + wco * cm);
      const float hv  = ogv * tanhf(cm);
      out[idx] = cm;
      out[(size_t)BATCH * OUT_F + idx] = hv;
    }
  }
}

// ---------------------------------------------------------------------------
extern "C" void kernel_launch(void* const* d_in, const int* in_sizes, int n_in,
                              void* d_out, int out_size, void* d_ws, size_t ws_size,
                              hipStream_t stream) {
  (void)in_sizes; (void)n_in; (void)out_size;
  const float* x       = (const float*)d_in[0];
  const float* C       = (const float*)d_in[1];
  const float* H       = (const float*)d_in[2];
  const float* fg_w_c  = (const float*)d_in[3];
  const float* fg_w_h  = (const float*)d_in[4];
  const float* fg_w_x  = (const float*)d_in[5];
  const float* fg_b    = (const float*)d_in[6];
  const float* ig_w_c  = (const float*)d_in[7];
  const float* ig_w_h  = (const float*)d_in[8];
  const float* ig_w_x  = (const float*)d_in[9];
  const float* ig_b    = (const float*)d_in[10];
  const float* in_w_h  = (const float*)d_in[11];
  const float* in_w_x  = (const float*)d_in[12];
  const float* in_b    = (const float*)d_in[13];
  const float* og_w_cn = (const float*)d_in[14];
  const float* og_w_h  = (const float*)d_in[15];
  const float* og_w_x  = (const float*)d_in[16];
  const float* og_b    = (const float*)d_in[17];
  const int*   nstep   = (const int*)d_in[18];
  float* out = (float*)d_out;

  const size_t A_elems = (size_t)BATCH * KTOT;            // 6,291,456
  const size_t B_elems = (size_t)4 * OUT_F * KTOT;        // 1,572,864
  const size_t ws_needed = (A_elems + B_elems) * sizeof(unsigned short);

  if (ws_size >= ws_needed) {
    unsigned short* Abf = (unsigned short*)d_ws;
    unsigned short* Bbf = Abf + A_elems;
    const int pack_blocks = (int)((A_elems + B_elems) / 8 / 256);  // 3840
    pack_bf16_kernel<<<pack_blocks, 256, 0, stream>>>(
        x, H, fg_w_h, fg_w_x, ig_w_h, ig_w_x, in_w_h, in_w_x,
        og_w_h, og_w_x, nstep, Abf, Bbf);
    const int gemm_blocks = (BATCH / 128) * (OUT_F / 16);          // 2048
    lstm_gemm_bf16<<<gemm_blocks, 256, 0, stream>>>(
        Abf, Bbf, C, fg_w_c, fg_b, ig_w_c, ig_b, in_b, og_w_cn, og_b, out);
  } else {
    const int blocks = (BATCH / 32) * (OUT_F / 16) / 8;            // 1024
    lstm_wmma_f32_fallback<<<blocks, 256, 0, stream>>>(
        x, C, H, fg_w_c, fg_w_h, fg_w_x, fg_b, ig_w_c, ig_w_h, ig_w_x, ig_b,
        in_w_h, in_w_x, in_b, og_w_cn, og_w_h, og_w_x, og_b, nstep, out);
  }
}
